// MultiHeadAttention_71794673320155
// MI455X (gfx1250) — matrix operand order repaired and checked by reading, not compiled, not run on hardware
//
#include <hip/hip_runtime.h>
#include <hip/hip_bf16.h>

typedef __bf16 bf16;
typedef bf16  v16bf __attribute__((ext_vector_type(16)));
typedef bf16  v4bf  __attribute__((ext_vector_type(4)));
typedef float v8f   __attribute__((ext_vector_type(8)));

#define LOG2E 1.4426950408889634f

// ---------------------------------------------------------------------------
// Kernel 1: fused QKV projection.  Y = X(8192x1024 f32) @ W(1024x1024 f32),
// bf16 on the fly; Q pre-scaled by 1/8 (exact).  K-chunk = 64.
// grid = (N/256=4, M/64=128, 3), block = 256 (8 waves: 2x4, 32x64 per wave)
// LDS: A row-major [64][64], B transposed [256][64] (n-major -> contiguous
// B-fragment gathers).  1.5 ds_load_b128 per WMMA.
// ---------------------------------------------------------------------------
__global__ __launch_bounds__(256) void qkv_gemm_kernel(
    const float* __restrict__ X,
    const float* __restrict__ Wq, const float* __restrict__ Wk,
    const float* __restrict__ Wv,
    bf16* __restrict__ Q, bf16* __restrict__ Kp, bf16* __restrict__ Vp)
{
    __shared__ bf16 als[64][64];     // A tile  (M x K), row-major
    __shared__ bf16 blsT[256][64];   // B tile  transposed: [n][k]

    const int z = blockIdx.z;
    const float* W = (z == 0) ? Wq : ((z == 1) ? Wk : Wv);
    bf16* Y        = (z == 0) ? Q  : ((z == 1) ? Kp : Vp);
    const float oscale = (z == 0) ? 0.125f : 1.0f;

    const int m0   = blockIdx.y * 64;
    const int n0   = blockIdx.x * 256;
    const int t    = threadIdx.x;
    const int wid  = t >> 5;
    const int lane = t & 31;
    const int wM   = wid >> 2;       // 0..1  -> 32 rows each
    const int wN   = wid & 3;        // 0..3  -> 64 cols each
    const int hh   = lane >> 4;
    const int l16  = lane & 15;

    v8f acc[2][4] = {};

    for (int k0 = 0; k0 < 1024; k0 += 64) {
        __syncthreads();
        if (k0 + 64 < 1024)
            __builtin_prefetch(&W[(size_t)(k0 + 64) * 1024 + n0 + (t & 255)], 0, 1);
        // stage A: 64x64 f32 -> bf16; float4 loads, packed b64 LDS stores
        #pragma unroll
        for (int i = 0; i < 4; ++i) {
            int u = t + i * 256;               // 1024 float4 units
            int r = u >> 4, cq = (u & 15) * 4;
            float4 f = *reinterpret_cast<const float4*>(
                &X[(size_t)(m0 + r) * 1024 + k0 + cq]);
            v4bf v; v[0] = (bf16)f.x; v[1] = (bf16)f.y;
                    v[2] = (bf16)f.z; v[3] = (bf16)f.w;
            *reinterpret_cast<v4bf*>(&als[r][cq]) = v;
        }
        // stage B transposed: each thread loads 4 rows of one column
        // (coalesced across lanes), packs, one b64 LDS store
        #pragma unroll
        for (int i = 0; i < 16; ++i) {
            int u = t + i * 256;               // 4096 quad units
            int c = u & 255, rq = (u >> 8) * 4;
            v4bf v;
            #pragma unroll
            for (int j = 0; j < 4; ++j)
                v[j] = (bf16)W[(size_t)(k0 + rq + j) * 1024 + n0 + c];
            *reinterpret_cast<v4bf*>(&blsT[c][rq]) = v;
        }
        __syncthreads();

        #pragma unroll
        for (int kc = 0; kc < 2; ++kc) {
            v16bf afr[2], bfr[4];
            #pragma unroll
            for (int mt = 0; mt < 2; ++mt) {
                int mrow = wM * 32 + mt * 16 + l16;
                #pragma unroll
                for (int e = 0; e < 16; ++e) {
                    int kk = 16 * hh + e;  // A-layout
                    afr[mt][e] = als[mrow][kc * 32 + kk];
                }
            }
            #pragma unroll
            for (int nt = 0; nt < 4; ++nt) {
                int ncol = wN * 64 + nt * 16 + l16;
                #pragma unroll
                for (int e = 0; e < 16; ++e)                             // B-layout
                    bfr[nt][e] = blsT[ncol][kc * 32 + 16 * hh + e];      // contiguous
            }
            #pragma unroll
            for (int mt = 0; mt < 2; ++mt)
                #pragma unroll
                for (int nt = 0; nt < 4; ++nt)
                    acc[mt][nt] = __builtin_amdgcn_wmma_f32_16x16x32_bf16(
                        false, afr[mt], false, bfr[nt], (short)0, acc[mt][nt],
                        false, false);
        }
    }

    #pragma unroll
    for (int mt = 0; mt < 2; ++mt)
        #pragma unroll
        for (int nt = 0; nt < 4; ++nt)
            #pragma unroll
            for (int r = 0; r < 8; ++r) {
                int m = m0 + wM * 32 + mt * 16 + r + 8 * hh;             // C/D layout
                int n = n0 + wN * 64 + nt * 16 + l16;
                Y[(size_t)m * 1024 + n] = (bf16)(acc[mt][nt][r] * oscale);
            }
}

// ---------------------------------------------------------------------------
// Kernel 2: causal flash attention over bf16 Q/K/V (Q pre-scaled).
// grid = (T/128=16, H=16, B=4), block = 256. Each wave owns 16 queries.
// K tile row-major (S-gemm gathers along contiguous features);
// V tile transposed [feat][key] (PV-gemm gathers along contiguous keys).
// ---------------------------------------------------------------------------
__global__ __launch_bounds__(256) void attn_kernel(
    const bf16* __restrict__ Q, const bf16* __restrict__ Kp,
    const bf16* __restrict__ Vp, bf16* __restrict__ CTX)
{
    __shared__ bf16 kls[32][64];      // [key][feat]
    __shared__ bf16 vlsT[64][32];     // [feat][key]
    __shared__ bf16 pls[8][16][32];   // per-wave P staging

    const int qblk = blockIdx.x;
    const int h    = blockIdx.y;
    const int b    = blockIdx.z;
    const int t    = threadIdx.x;
    const int wid  = t >> 5;
    const int lane = t & 31;
    const int hh   = lane >> 4;
    const int l16  = lane & 15;
    const int q0   = qblk * 128 + wid * 16;
    const size_t rowbase = (size_t)b * 2048;

    // gather this wave's Q A-fragments (contiguous 8-element runs -> b128 loads)
    v16bf qfr[2];
    #pragma unroll
    for (int kc = 0; kc < 2; ++kc)
        #pragma unroll
        for (int e = 0; e < 16; ++e) {
            int kk = 16 * hh + e;
            qfr[kc][e] = Q[(rowbase + q0 + l16) * 1024 + h * 64 + kc * 32 + kk];
        }

    float mi[8], li[8];
    v8f acc[4] = {};
    #pragma unroll
    for (int r = 0; r < 8; ++r) { mi[r] = -1e30f; li[r] = 0.f; }

    const int kb_end = (qblk * 128 + 127) >> 5;   // WG-uniform
    const int kb_my  = (q0 + 15) >> 5;            // wave-uniform

    for (int kb = 0; kb <= kb_end; ++kb) {
        __syncthreads();
        // stage K row-major: 8-byte vector copies
        #pragma unroll
        for (int i = 0; i < 2; ++i) {
            int u = t + i * 256;                  // 512 quad units
            int r = u >> 4, cq = (u & 15) * 4;
            *reinterpret_cast<v4bf*>(&kls[r][cq]) =
                *reinterpret_cast<const v4bf*>(
                    &Kp[(rowbase + kb * 32 + r) * 1024 + h * 64 + cq]);
        }
        // stage V transposed: 4 rows of one feature column per unit
        #pragma unroll
        for (int i = 0; i < 2; ++i) {
            int u = t + i * 256;                  // 512 quad units
            int c = u & 63, rq = (u >> 6) * 4;
            v4bf v;
            #pragma unroll
            for (int j = 0; j < 4; ++j)
                v[j] = Vp[(rowbase + kb * 32 + rq + j) * 1024 + h * 64 + c];
            *reinterpret_cast<v4bf*>(&vlsT[c][rq]) = v;
        }
        __syncthreads();
        if (kb > kb_my) continue;                 // wave-uniform: EXEC stays full

        // S(16x32) = Q(16x64) * K^T : 4 WMMA
        v8f st[2] = {};
        #pragma unroll
        for (int tt = 0; tt < 2; ++tt)
            #pragma unroll
            for (int kc = 0; kc < 2; ++kc) {
                v16bf bfr;
                #pragma unroll
                for (int e = 0; e < 16; ++e)                       // contiguous
                    bfr[e] = kls[tt * 16 + l16][kc * 32 + 16 * hh + e];
                st[tt] = __builtin_amdgcn_wmma_f32_16x16x32_bf16(
                    false, qfr[kc], false, bfr, (short)0, st[tt], false, false);
            }

        // causal mask + online softmax (fp32, C/D row striping)
        float scl[8], p[2][8];
        #pragma unroll
        for (int r = 0; r < 8; ++r) {
            int q = q0 + r + 8 * hh;
            float mb = -1e30f;
            #pragma unroll
            for (int tt = 0; tt < 2; ++tt) {
                int key = kb * 32 + tt * 16 + l16;
                float s = st[tt][r];
                if (key > q) s = -1e30f;
                st[tt][r] = s;
                mb = fmaxf(mb, s);
            }
            #pragma unroll
            for (int off = 1; off < 16; off <<= 1)
                mb = fmaxf(mb, __shfl_xor(mb, off, 32));
            float mnew = fmaxf(mi[r], mb);
            scl[r] = exp2f((mi[r] - mnew) * LOG2E);
            float prsum = 0.f;
            #pragma unroll
            for (int tt = 0; tt < 2; ++tt) {
                float pv = exp2f((st[tt][r] - mnew) * LOG2E);
                p[tt][r] = pv;
                prsum += pv;
            }
            #pragma unroll
            for (int off = 1; off < 16; off <<= 1)
                prsum += __shfl_xor(prsum, off, 32);
            li[r] = li[r] * scl[r] + prsum;
            mi[r] = mnew;
        }

        // D-layout -> LDS -> A-layout for P
        #pragma unroll
        for (int tt = 0; tt < 2; ++tt)
            #pragma unroll
            for (int r = 0; r < 8; ++r)
                pls[wid][r + 8 * hh][tt * 16 + l16] = (bf16)p[tt][r];

        #pragma unroll
        for (int nt = 0; nt < 4; ++nt)
            #pragma unroll
            for (int r = 0; r < 8; ++r)
                acc[nt][r] *= scl[r];

        v16bf pfr;
        #pragma unroll
        for (int e = 0; e < 16; ++e) {
            int kk = 16 * hh + e;
            pfr[e] = pls[wid][l16][kk];
        }
        #pragma unroll
        for (int nt = 0; nt < 4; ++nt) {          // O += P(16x32) * V(32x64)
            v16bf bfr;
            #pragma unroll
            for (int e = 0; e < 16; ++e)                           // contiguous
                bfr[e] = vlsT[nt * 16 + l16][16 * hh + e];
            acc[nt] = __builtin_amdgcn_wmma_f32_16x16x32_bf16(
                false, pfr, false, bfr, (short)0, acc[nt], false, false);
        }
    }

    // normalize rows and emit ctx in (B, T, H*64) bf16 layout
    #pragma unroll
    for (int nt = 0; nt < 4; ++nt)
        #pragma unroll
        for (int r = 0; r < 8; ++r) {
            int m = q0 + r + 8 * hh;
            int n = h * 64 + nt * 16 + l16;
            CTX[(rowbase + m) * 1024 + n] = (bf16)(acc[nt][r] / li[r]);
        }
}

// ---------------------------------------------------------------------------
// Kernel 3: output projection.  OUT(f32) = CTX(bf16) @ Wo(f32->bf16) + bo
// grid = (4, 128), block = 256.  Same tiling as kernel 1; A is already bf16.
// ---------------------------------------------------------------------------
__global__ __launch_bounds__(256) void out_proj_kernel(
    const bf16* __restrict__ CTX, const float* __restrict__ Wo,
    const float* __restrict__ bo, float* __restrict__ OUT)
{
    __shared__ bf16 als[64][64];
    __shared__ bf16 blsT[256][64];

    const int m0   = blockIdx.y * 64;
    const int n0   = blockIdx.x * 256;
    const int t    = threadIdx.x;
    const int wid  = t >> 5;
    const int lane = t & 31;
    const int wM   = wid >> 2;
    const int wN   = wid & 3;
    const int hh   = lane >> 4;
    const int l16  = lane & 15;

    v8f acc[2][4] = {};

    for (int k0 = 0; k0 < 1024; k0 += 64) {
        __syncthreads();
        if (k0 + 64 < 1024)
            __builtin_prefetch(&Wo[(size_t)(k0 + 64) * 1024 + n0 + (t & 255)], 0, 1);
        // stage A (bf16 source): 8-byte vector copies
        #pragma unroll
        for (int i = 0; i < 4; ++i) {
            int u = t + i * 256;                  // 1024 quad units
            int r = u >> 4, cq = (u & 15) * 4;
            *reinterpret_cast<v4bf*>(&als[r][cq]) =
                *reinterpret_cast<const v4bf*>(
                    &CTX[(size_t)(m0 + r) * 1024 + k0 + cq]);
        }
        // stage B transposed
        #pragma unroll
        for (int i = 0; i < 16; ++i) {
            int u = t + i * 256;
            int c = u & 255, rq = (u >> 8) * 4;
            v4bf v;
            #pragma unroll
            for (int j = 0; j < 4; ++j)
                v[j] = (bf16)Wo[(size_t)(k0 + rq + j) * 1024 + n0 + c];
            *reinterpret_cast<v4bf*>(&blsT[c][rq]) = v;
        }
        __syncthreads();

        #pragma unroll
        for (int kc = 0; kc < 2; ++kc) {
            v16bf afr[2], bfr[4];
            #pragma unroll
            for (int mt = 0; mt < 2; ++mt) {
                int mrow = wM * 32 + mt * 16 + l16;
                #pragma unroll
                for (int e = 0; e < 16; ++e) {
                    int kk = 16 * hh + e;
                    afr[mt][e] = als[mrow][kc * 32 + kk];
                }
            }
            #pragma unroll
            for (int nt = 0; nt < 4; ++nt) {
                int ncol = wN * 64 + nt * 16 + l16;
                #pragma unroll
                for (int e = 0; e < 16; ++e)
                    bfr[nt][e] = blsT[ncol][kc * 32 + 16 * hh + e];
            }
            #pragma unroll
            for (int mt = 0; mt < 2; ++mt)
                #pragma unroll
                for (int nt = 0; nt < 4; ++nt)
                    acc[mt][nt] = __builtin_amdgcn_wmma_f32_16x16x32_bf16(
                        false, afr[mt], false, bfr[nt], (short)0, acc[mt][nt],
                        false, false);
        }
    }

    #pragma unroll
    for (int mt = 0; mt < 2; ++mt)
        #pragma unroll
        for (int nt = 0; nt < 4; ++nt)
            #pragma unroll
            for (int r = 0; r < 8; ++r) {
                int m = m0 + wM * 32 + mt * 16 + r + 8 * hh;
                int n = n0 + wN * 64 + nt * 16 + l16;
                OUT[(size_t)m * 1024 + n] = acc[mt][nt][r] + bo[n];
            }
}

// ---------------------------------------------------------------------------
extern "C" void kernel_launch(void* const* d_in, const int* in_sizes, int n_in,
                              void* d_out, int out_size, void* d_ws, size_t ws_size,
                              hipStream_t stream) {
    const float* x  = (const float*)d_in[0];
    const float* Wq = (const float*)d_in[1];
    const float* Wk = (const float*)d_in[2];
    const float* Wv = (const float*)d_in[3];
    const float* Wo = (const float*)d_in[4];
    const float* bo = (const float*)d_in[5];
    float* out = (float*)d_out;

    // workspace: Q, K, V, CTX as bf16 [8192 x 1024] each (16 MB apiece)
    const size_t SZ = (size_t)8192 * 1024 * sizeof(bf16);
    char* ws = (char*)d_ws;
    bf16* Q   = (bf16*)(ws);
    bf16* Kp  = (bf16*)(ws + SZ);
    bf16* Vp  = (bf16*)(ws + 2 * SZ);
    bf16* CTX = (bf16*)(ws + 3 * SZ);

    dim3 gQKV(4, 128, 3);
    qkv_gemm_kernel<<<gQKV, 256, 0, stream>>>(x, Wq, Wk, Wv, Q, Kp, Vp);

    dim3 gATT(16, 16, 4);
    attn_kernel<<<gATT, 256, 0, stream>>>(Q, Kp, Vp, CTX);

    dim3 gOUT(4, 128, 1);
    out_proj_kernel<<<gOUT, 256, 0, stream>>>(CTX, Wo, bo, out);
}
